// GCN_47150150976174
// MI455X (gfx1250) — compile-verified, matrix operand order checked by reading, not run on hardware
//
#include <hip/hip_runtime.h>

typedef __attribute__((ext_vector_type(16))) _Float16 v16h;
typedef __attribute__((ext_vector_type(8)))  _Float16 h8;
typedef __attribute__((ext_vector_type(8)))  float    v8f;

#define F    128
#define HIDN 128
#define GNUM 64
#define CNUM 8
#define EPSV 1e-5f

// ----------------------------- init kernels -----------------------------

__global__ void k_zero_f32(float* __restrict__ p, int n) {
  int i = blockIdx.x * blockDim.x + threadIdx.x;
  int st = gridDim.x * blockDim.x;
  for (; i < n; i += st) p[i] = 0.0f;
}

__global__ void k_fill_u32(unsigned* __restrict__ p, unsigned v, int n) {
  int i = blockIdx.x * blockDim.x + threadIdx.x;
  int st = gridDim.x * blockDim.x;
  for (; i < n; i += st) p[i] = v;
}

// ----------------------------- batchnorm --------------------------------

// blockDim = 128; thread t owns feature column t. Coalesced row reads.
__global__ void k_bn_stats(const float* __restrict__ x,
                           float* __restrict__ sum, float* __restrict__ sumsq,
                           int N) {
  int t = threadIdx.x;
  float s = 0.0f, q = 0.0f;
  for (int r = blockIdx.x; r < N; r += gridDim.x) {
    float v = x[r * F + t];
    s += v;
    q += v * v;
  }
  unsafeAtomicAdd(&sum[t], s);
  unsafeAtomicAdd(&sumsq[t], q);
}

// Normalize and write f16 copy for the WMMA GEMM.
__global__ void k_bn_apply(const float* __restrict__ x,
                           const float* __restrict__ sum, const float* __restrict__ sumsq,
                           const float* __restrict__ bnw, const float* __restrict__ bnb,
                           _Float16* __restrict__ xh, int N) {
  int i = blockIdx.x * blockDim.x + threadIdx.x;
  int st = gridDim.x * blockDim.x;
  int total = N * F;
  float invN = 1.0f / (float)N;
  for (; i < total; i += st) {
    int f = i & (F - 1);
    float mu  = sum[f] * invN;
    float var = sumsq[f] * invN - mu * mu;
    float rs  = rsqrtf(var + EPSV);
    float v   = (x[i] - mu) * rs * bnw[f] + bnb[f];
    xh[i] = (_Float16)v;
  }
}

// ----------------------------- W1 fragment pack -------------------------
// Pack W1 [K=128, N=128] (row major) into WMMA B-fragment order:
// bfrag[((nt*4 + kc)*32 + lane)*16 + j], per ISA 16-bit B layout:
//   lane<16 : k = kcb + {0..7, 16..23};  lane>=16 : k = kcb + {8..15, 24..31}
__global__ void k_w1_pack(const float* __restrict__ W1, _Float16* __restrict__ bfrag) {
  int i = blockIdx.x * blockDim.x + threadIdx.x;   // 8*4*32*16 = 16384
  if (i >= 8 * 4 * 32 * 16) return;
  int j    = i & 15;
  int lane = (i >> 4) & 31;
  int kc   = (i >> 9) & 3;
  int nt   = i >> 11;
  int koff = ((lane >> 4) & 1) * 8;
  int k = kc * 32 + ((j < 8) ? (koff + j) : (16 + koff + (j - 8)));
  int n = nt * 16 + (lane & 15);
  bfrag[i] = (_Float16)W1[k * HIDN + n];
}

// ----------------------------- GEMM (WMMA) ------------------------------
// One wave32 computes a 16x128 strip of h0 = xh @ W1.  K=128 in 4 chunks of
// 32; A fragment reused across all 8 N-tiles (32 v_wmma per wave).
__global__ void k_gemm(const _Float16* __restrict__ xh,
                       const _Float16* __restrict__ bfrag,
                       float* __restrict__ h0, int mtiles) {
  int wave = (int)((blockIdx.x * blockDim.x + threadIdx.x) >> 5);
  int lane = threadIdx.x & 31;
  if (wave >= mtiles) return;                 // whole-wave exit: EXEC stays all-1s

  int m0   = wave * 16;
  int mrow = m0 + (lane & 15);
  int koff = ((lane >> 4) & 1) * 8;

  v8f acc[8] = {};
#pragma unroll
  for (int kc = 0; kc < 4; ++kc) {
    const _Float16* base = xh + mrow * F + kc * 32 + koff;
    h8 lo = *(const h8*)(base);               // K = kcb + koff + 0..7
    h8 hi = *(const h8*)(base + 16);          // K = kcb + 16 + koff + 0..7
    v16h a = __builtin_shufflevector(lo, hi, 0, 1, 2, 3, 4, 5, 6, 7,
                                             8, 9, 10, 11, 12, 13, 14, 15);
#pragma unroll
    for (int nt = 0; nt < 8; ++nt) {
      v16h b = *(const v16h*)(bfrag + (size_t)((nt * 4 + kc) * 32 + lane) * 16);
      acc[nt] = __builtin_amdgcn_wmma_f32_16x16x32_f16(
          false, a, false, b, (short)0, acc[nt], false, false);
    }
  }

  int rowhi = ((lane >> 4) & 1) * 8;
  int col0  = lane & 15;
#pragma unroll
  for (int nt = 0; nt < 8; ++nt) {
#pragma unroll
    for (int r = 0; r < 8; ++r) {
      h0[(m0 + r + rowhi) * HIDN + nt * 16 + col0] = acc[nt][r];
    }
  }
}

// ----------------------------- graph ops --------------------------------

__global__ void k_deg(const int* __restrict__ col, unsigned* __restrict__ deg, int E) {
  int i = blockIdx.x * blockDim.x + threadIdx.x;
  int st = gridDim.x * blockDim.x;
  for (; i < E; i += st) atomicAdd(&deg[col[i]], 1u);
}

__global__ void k_counts(const int* __restrict__ batch, unsigned* __restrict__ counts, int N) {
  int i = blockIdx.x * blockDim.x + threadIdx.x;
  int st = gridDim.x * blockDim.x;
  for (; i < N; i += st) atomicAdd(&counts[batch[i]], 1u);
}

// One wave32 per edge: lane handles 4 consecutive features (float4 gather,
// 4 native f32 global atomics to the target row).
__global__ void k_scatter(const int* __restrict__ row, const int* __restrict__ col,
                          const unsigned* __restrict__ deg,
                          const float* __restrict__ h0, float* __restrict__ hagg, int E) {
  int wave = (int)((blockIdx.x * blockDim.x + threadIdx.x) >> 5);
  int lane = threadIdx.x & 31;
  int nw = (gridDim.x * blockDim.x) >> 5;
  for (int e = wave; e < E; e += nw) {
    int r = row[e], c = col[e];
    float w = rsqrtf((float)deg[r]) * rsqrtf((float)deg[c]);
    float4 v = *(const float4*)(h0 + (size_t)r * HIDN + lane * 4);
    float* dst = hagg + (size_t)c * HIDN + lane * 4;
    unsafeAtomicAdd(dst + 0, v.x * w);
    unsafeAtomicAdd(dst + 1, v.y * w);
    unsafeAtomicAdd(dst + 2, v.z * w);
    unsafeAtomicAdd(dst + 3, v.w * w);
  }
}

// Add self-loop term + bias, ReLU; accumulate global LN stats and per-graph
// pooled sums (LN affine is deferred to the final kernel since pooling is
// linear and LN(graph mode) is a global scalar affine).
__global__ void k_finalize(const float* __restrict__ hagg, const float* __restrict__ h0,
                           const unsigned* __restrict__ deg, const float* __restrict__ cbias,
                           const int* __restrict__ batch,
                           float* __restrict__ pool, float* __restrict__ lnacc, int N) {
  __shared__ float s1[256];
  __shared__ float s2[256];
  int tid = threadIdx.x;
  int i = blockIdx.x * blockDim.x + tid;
  int st = gridDim.x * blockDim.x;
  int total = N * HIDN;
  float ls = 0.0f, lq = 0.0f;
  for (; i < total; i += st) {
    int n = i >> 7;
    int f = i & (HIDN - 1);
    float v = hagg[i] + h0[i] / (float)deg[n] + cbias[f];
    v = fmaxf(v, 0.0f);
    ls += v;
    lq += v * v;
    unsafeAtomicAdd(&pool[(size_t)batch[n] * HIDN + f], v);
  }
  s1[tid] = ls;
  s2[tid] = lq;
  __syncthreads();
  for (int o = 128; o > 0; o >>= 1) {
    if (tid < o) { s1[tid] += s1[tid + o]; s2[tid] += s2[tid + o]; }
    __syncthreads();
  }
  if (tid == 0) {
    unsafeAtomicAdd(&lnacc[0], s1[0]);
    unsafeAtomicAdd(&lnacc[1], s2[0]);
  }
}

// LN affine on pooled [G,128] + classifier head -> out [G,8].
__global__ void k_final(const float* __restrict__ pool, const unsigned* __restrict__ counts,
                        const float* __restrict__ lnacc,
                        const float* __restrict__ lnw, const float* __restrict__ lnb,
                        const float* __restrict__ linW, const float* __restrict__ linb,
                        float* __restrict__ out, int N) {
  int tid = threadIdx.x;                       // 512 = 64 graphs x 8 classes
  int g = tid >> 3;
  int c = tid & 7;
  float invT = 1.0f / ((float)N * (float)HIDN);
  float mean = lnacc[0] * invT;
  float var  = lnacc[1] * invT - mean * mean;
  float rstd = rsqrtf(var + EPSV);
  float inv  = 1.0f / fmaxf((float)counts[g], 1.0f);
  float acc = 0.0f;
  for (int f = 0; f < HIDN; ++f) {
    float gp = pool[(size_t)g * HIDN + f] * inv;
    float hl = (gp - mean) * rstd * lnw[f] + lnb[f];
    acc += hl * linW[c * HIDN + f];
  }
  out[g * CNUM + c] = acc + linb[c];
}

// ----------------------------- launcher ---------------------------------

extern "C" void kernel_launch(void* const* d_in, const int* in_sizes, int n_in,
                              void* d_out, int out_size, void* d_ws, size_t ws_size,
                              hipStream_t stream) {
  (void)n_in; (void)out_size; (void)ws_size;
  const float* x     = (const float*)d_in[0];
  const int*   ei    = (const int*)d_in[1];
  const int*   batch = (const int*)d_in[2];
  const float* bnw   = (const float*)d_in[3];
  const float* bnb   = (const float*)d_in[4];
  const float* W1    = (const float*)d_in[5];
  const float* cbias = (const float*)d_in[6];
  const float* lnw   = (const float*)d_in[7];
  const float* lnb   = (const float*)d_in[8];
  const float* linW  = (const float*)d_in[9];
  const float* linb  = (const float*)d_in[10];
  float* out = (float*)d_out;

  int N = in_sizes[0] / F;
  int E = in_sizes[1] / 2;
  const int* row = ei;       // edge_index[0]
  const int* col = ei + E;   // edge_index[1]

  // ---- workspace layout (256B-aligned regions) ----
  char* ws = (char*)d_ws;
  size_t off = 0;
  auto carve = [&](size_t bytes) -> void* {
    void* p = ws + off;
    off += (bytes + 255) & ~(size_t)255;
    return p;
  };
  float*    h0    = (float*)carve(sizeof(float) * (size_t)N * HIDN);     // 51.2 MB
  float*    hagg  = (float*)carve(sizeof(float) * (size_t)N * HIDN);     // 51.2 MB
  _Float16* xh    = (_Float16*)hagg;   // alias: xh dead before hagg is zeroed
  unsigned* deg   = (unsigned*)carve(sizeof(unsigned) * (size_t)N);
  float*    bnsum = (float*)carve(sizeof(float) * 2 * F);
  float*    bnsq  = bnsum + F;
  float*    lnacc = (float*)carve(sizeof(float) * 2);
  float*    pool  = (float*)carve(sizeof(float) * GNUM * HIDN);
  unsigned* cnts  = (unsigned*)carve(sizeof(unsigned) * GNUM);
  _Float16* bfrag = (_Float16*)carve(sizeof(_Float16) * 8 * 4 * 32 * 16);

  // ---- init accumulators ----
  k_zero_f32<<<1, 256, 0, stream>>>(bnsum, 2 * F);
  k_zero_f32<<<1, 64, 0, stream>>>(lnacc, 2);
  k_zero_f32<<<32, 256, 0, stream>>>(pool, GNUM * HIDN);
  k_fill_u32<<<1, 64, 0, stream>>>(cnts, 0u, GNUM);
  k_fill_u32<<<196, 256, 0, stream>>>(deg, 1u, N);   // self-loop degree

  // ---- batchnorm ----
  k_bn_stats<<<512, 128, 0, stream>>>(x, bnsum, bnsq, N);
  k_bn_apply<<<2048, 256, 0, stream>>>(x, bnsum, bnsq, bnw, bnb, xh, N);

  // ---- GEMM (WMMA) ----
  k_w1_pack<<<64, 256, 0, stream>>>(W1, bfrag);
  int mtiles = N / 16;                                // N = 100000 -> 6250
  int gblocks = (mtiles * 32 + 255) / 256;
  k_gemm<<<gblocks, 256, 0, stream>>>(xh, bfrag, h0, mtiles);

  // ---- scatter-add over edges ----
  k_zero_f32<<<2048, 256, 0, stream>>>(hagg, N * HIDN);  // xh dead now
  k_deg<<<1024, 256, 0, stream>>>(col, deg, E);
  k_counts<<<512, 256, 0, stream>>>(batch, cnts, N);
  k_scatter<<<4096, 256, 0, stream>>>(row, col, deg, h0, hagg, E);

  // ---- bias + relu + LN stats + pooling ----
  k_finalize<<<2048, 256, 0, stream>>>(hagg, h0, deg, cbias, batch, pool, lnacc, N);

  // ---- LN affine + classifier ----
  k_final<<<1, 512, 0, stream>>>(pool, cnts, lnacc, lnw, lnb, linW, linb, out, N);
}
